// I64Attention_25323127177577
// MI455X (gfx1250) — matrix operand order repaired and checked by reading, not compiled, not run on hardware
//
#include <hip/hip_runtime.h>

// ---------------------------------------------------------------------------
// Types for CDNA5 WMMA (wave32): v_wmma_f32_16x16x32_bf16
// ---------------------------------------------------------------------------
typedef __attribute__((ext_vector_type(16))) __bf16 v16bf;
typedef __attribute__((ext_vector_type(8)))  float  v8f;

union BfOp {
  v16bf v;
  uint4 u[2];
};

__device__ __forceinline__ unsigned short f2bf(float f) {
  unsigned int u = __float_as_uint(f);
  unsigned int r = (u + 0x7FFFu + ((u >> 16) & 1u)) >> 16;  // round-nearest-even
  return (unsigned short)r;
}

// gfx1250 async global->LDS copy (ASYNCcnt-tracked). Per-lane 16 bytes.
// Inline asm (bypasses builtin-arity differences between toolchains).
__device__ __forceinline__ void async_copy_b128(unsigned lds_byte_off,
                                                const unsigned short* g) {
  asm volatile("global_load_async_to_lds_b128 %0, %1, off"
               :: "v"(lds_byte_off), "v"((unsigned long long)(size_t)g)
               : "memory");
}
__device__ __forceinline__ void wait_async_all() {
  asm volatile("s_wait_asynccnt 0" ::: "memory");
}
// Low 32 bits of a generic shared-pointer = LDS byte offset.
__device__ __forceinline__ unsigned lds_off(const void* p) {
  return (unsigned)(size_t)p;
}

#define SEQ   2048
#define HDIM  128
#define NHQ   16
#define NKVH  4

// ---------------------------------------------------------------------------
// f32 -> bf16 bulk conversion (vectorized), done ONCE per tensor.
// ---------------------------------------------------------------------------
__global__ __launch_bounds__(256) void to_bf16_kernel(
    const float* __restrict__ in, unsigned short* __restrict__ out, int n)
{
  int i = (blockIdx.x * 256 + threadIdx.x) * 4;
  if (i >= n) return;
  float4 f = *(const float4*)(in + i);
  uint2 o;
  o.x = (unsigned)f2bf(f.x) | ((unsigned)f2bf(f.y) << 16);
  o.y = (unsigned)f2bf(f.z) | ((unsigned)f2bf(f.w) << 16);
  *(uint2*)(out + i) = o;
}

// ---------------------------------------------------------------------------
// GEMM: C[M,N] = A1[M,K] @ W1[N,K]^T (+ A2 @ W2^T if A2 != null), bf16 in,
// f32 out. Block tile 64(M) x 256(N); 8 waves as 2(M) x 4(N); wave tile 32x64.
// Double-buffered LDS filled with global_load_async_to_lds_b128.
// ---------------------------------------------------------------------------
__device__ __forceinline__ void stage_tile(
    const unsigned short* __restrict__ A, const unsigned short* __restrict__ W,
    int m0, int n0, int K, int k0, int t, unsigned asb, unsigned bsb)
{
  {
    int r = t >> 2, c = (t & 3) * 8;                 // A: 64x32 bf16, 1x b128/thr
    async_copy_b128(asb + t * 16, A + (size_t)(m0 + r) * K + k0 + c);
  }
  #pragma unroll
  for (int i = 0; i < 4; ++i) {                      // W: 256x32 bf16, 4x b128/thr
    int tt = t + i * 256;
    int r = tt >> 2, c = (tt & 3) * 8;
    async_copy_b128(bsb + tt * 16, W + (size_t)(n0 + r) * K + k0 + c);
  }
}

__global__ __launch_bounds__(256) void gemm_bf16_async(
    const unsigned short* __restrict__ A1, const unsigned short* __restrict__ W1,
    const unsigned short* __restrict__ A2, const unsigned short* __restrict__ W2,
    float* __restrict__ C, int M, int N, int K)
{
  __shared__ __align__(16) unsigned short As[2][64 * 32];
  __shared__ __align__(16) unsigned short Bs[2][256 * 32];

  const int t    = threadIdx.x;
  const int lane = t & 31;
  const int wave = t >> 5;
  const int wm   = wave >> 2;    // 0..1
  const int wn   = wave & 3;     // 0..3
  const int m0   = blockIdx.y * 64;
  const int n0   = blockIdx.x * 256;
  const int l15  = lane & 15;
  const int hi   = (lane < 16) ? 0 : 1;

  const unsigned asb[2] = { lds_off(&As[0][0]), lds_off(&As[1][0]) };
  const unsigned bsb[2] = { lds_off(&Bs[0][0]), lds_off(&Bs[1][0]) };

  v8f acc[2][4] = {};

  const int KT = (A2 ? 2 : 1) * K;   // logical K (dual-phase fused)

  // Prologue: stage tile 0.
  stage_tile(A1, W1, m0, n0, K, 0, t, asb[0], bsb[0]);
  wait_async_all();
  __syncthreads();

  for (int kk = 0; kk < KT; kk += 32) {
    const int bi = (kk >> 5) & 1;

    // Stage next tile into the other buffer while computing this one.
    const int kn = kk + 32;
    if (kn < KT) {
      const unsigned short* Ap = (kn < K) ? A1 : A2;
      const unsigned short* Wp = (kn < K) ? W1 : W2;
      stage_tile(Ap, Wp, m0, n0, K, (kn < K) ? kn : kn - K, t,
                 asb[bi ^ 1], bsb[bi ^ 1]);
    }

    // A operands: lane<16 -> K 0..7 & 16..23 ; lane>=16 -> K 8..15 & 24..31
    BfOp a[2];
    #pragma unroll
    for (int mt = 0; mt < 2; ++mt) {
      const unsigned short* ap =
          &As[bi][(wm * 32 + mt * 16 + l15) * 32 + (hi ? 8 : 0)];
      a[mt].u[0] = *(const uint4*)(ap);
      a[mt].u[1] = *(const uint4*)(ap + 16);
    }
    #pragma unroll
    for (int nt = 0; nt < 4; ++nt) {
      // B operand: lane<16 -> K 0..7 & 16..23 ; lane>=16 -> K 8..15 & 24..31
      BfOp b;
      const unsigned short* bp =
          &Bs[bi][(wn * 64 + nt * 16 + l15) * 32 + (hi ? 8 : 0)];
      b.u[0] = *(const uint4*)(bp);
      b.u[1] = *(const uint4*)(bp + 16);
      #pragma unroll
      for (int mt = 0; mt < 2; ++mt)
        acc[mt][nt] = __builtin_amdgcn_wmma_f32_16x16x32_bf16(
            false, a[mt].v, false, b.v, (short)0, acc[mt][nt], false, false);
    }

    wait_async_all();
    __syncthreads();
  }

  // C layout: VGPR r, lanes 0-15 -> row r ; lanes 16-31 -> row r+8
  #pragma unroll
  for (int mt = 0; mt < 2; ++mt) {
    #pragma unroll
    for (int nt = 0; nt < 4; ++nt) {
      #pragma unroll
      for (int r = 0; r < 8; ++r) {
        int row = m0 + wm * 32 + mt * 16 + r + hi * 8;
        int col = n0 + wn * 64 + nt * 16 + l15;
        C[(size_t)row * N + col] = acc[mt][nt][r];
      }
    }
  }
}

// ---------------------------------------------------------------------------
// Per-head RMSNorm + RoPE, f32 [b][s][h][d] -> bf16 [b][h][s][d].
// One wave per (b,s,h) row of 128. outscale folds 1/sqrt(HD) into Q.
// ---------------------------------------------------------------------------
__global__ __launch_bounds__(256) void norm_rope_kernel(
    const float* __restrict__ x, const float* __restrict__ w,
    const int* __restrict__ positions,
    unsigned short* __restrict__ out, int nheads, float outscale)
{
  const int lane = threadIdx.x & 31;
  const int wave = threadIdx.x >> 5;
  const int row  = blockIdx.x * 8 + wave;
  const int h    = row % nheads;
  const int sq   = (row / nheads) % SEQ;
  const int b    = row / (nheads * SEQ);
  const size_t ibase = ((size_t)(b * SEQ + sq) * nheads + h) * HDIM;
  const size_t obase = ((size_t)(b * nheads + h) * SEQ + sq) * HDIM;

  const int i0 = lane, i1 = lane + 32;
  float a0 = x[ibase + i0], b0 = x[ibase + i0 + 64];
  float a1 = x[ibase + i1], b1 = x[ibase + i1 + 64];

  float ssq = a0 * a0 + b0 * b0 + a1 * a1 + b1 * b1;
  #pragma unroll
  for (int m = 1; m < 32; m <<= 1) ssq += __shfl_xor(ssq, m, 32);
  const float rms = rsqrtf(ssq * (1.0f / 128.0f) + 1e-6f);
  const float pos = (float)positions[sq];

  #pragma unroll
  for (int p = 0; p < 2; ++p) {
    int   i  = p ? i1 : i0;
    float xa = p ? a1 : a0;
    float xb = p ? b1 : b0;
    float na = xa * rms * w[i];
    float nb = xb * rms * w[i + 64];
    float inv = __powf(10000.0f, -(float)i * (1.0f / 64.0f));
    float sn, cs;
    __sincosf(pos * inv, &sn, &cs);
    out[obase + i]      = f2bf((na * cs - nb * sn) * outscale);
    out[obase + i + 64] = f2bf((nb * cs + na * sn) * outscale);
  }
}

// ---------------------------------------------------------------------------
// V repack: f32 [b][s][hkv][d] -> bf16 [b][hkv][d][s]  (d-major for PV B-op)
// ---------------------------------------------------------------------------
__global__ __launch_bounds__(256) void v_pack_kernel(
    const float* __restrict__ v, unsigned short* __restrict__ out)
{
  int o = blockIdx.x * blockDim.x + threadIdx.x;   // 2^21 elements
  int sq = o & (SEQ - 1);
  int d  = (o >> 11) & (HDIM - 1);
  int h  = (o >> 18) & (NKVH - 1);
  int b  = o >> 20;
  size_t in = (size_t)(b * SEQ + sq) * (NKVH * HDIM) + h * HDIM + d;
  out[o] = f2bf(v[in]);
}

// ---------------------------------------------------------------------------
// Flash attention (causal, GQA). Wave handles 16 query rows x full HD=128.
// Q bf16 [b][16][S][128], K bf16 [b][4][S][128], V bf16 [b][4][128][S].
// Out bf16 [b][s][16*128] (feeds the output-projection GEMM directly).
// Softmax scale pre-folded into Q.
// ---------------------------------------------------------------------------
#define FLASH_NEG (-1.0e30f)

__global__ __launch_bounds__(128) void flash_kernel(
    const unsigned short* __restrict__ Q,
    const unsigned short* __restrict__ Km,
    const unsigned short* __restrict__ Vm,
    unsigned short* __restrict__ Out)
{
  __shared__ __align__(16) unsigned short Plds[4][16][32];   // per-wave P tile

  const int lane  = threadIdx.x & 31;
  const int wave  = threadIdx.x >> 5;
  const int b     = blockIdx.z;
  const int h     = blockIdx.y;
  const int hkv   = h >> 2;                         // GROUPS = 4
  const int q0    = blockIdx.x * 64 + wave * 16;
  const int l15   = lane & 15;
  const int hi    = (lane < 16) ? 0 : 1;

  // Q A-operands for the 4 k-steps of d (K=32 each), held in registers.
  BfOp qa[4];
  const unsigned short* qrow =
      Q + ((size_t)(b * NHQ + h) * SEQ + (q0 + l15)) * HDIM;
  #pragma unroll
  for (int ks = 0; ks < 4; ++ks) {
    int base = 32 * ks + (hi ? 8 : 0);
    qa[ks].u[0] = *(const uint4*)(qrow + base);
    qa[ks].u[1] = *(const uint4*)(qrow + base + 16);
  }

  v8f O[8] = {};
  float mrun[8], lrun[8];
  #pragma unroll
  for (int r = 0; r < 8; ++r) { mrun[r] = FLASH_NEG; lrun[r] = 0.0f; }

  const unsigned short* krow =
      Km + ((size_t)(b * NKVH + hkv) * SEQ + l15) * HDIM;
  const unsigned short* vrow =
      Vm + ((size_t)(b * NKVH + hkv) * HDIM + l15) * SEQ;

  const int ntiles = (q0 + 16 + 31) >> 5;           // 32-key tiles needed
  for (int kt = 0; kt < ntiles; ++kt) {
    const int kb = kt * 32;

    // ---- scores: two 16x16 tiles over 32 keys, K=HD=128 in 4 steps ----
    v8f s[2] = {};
    #pragma unroll
    for (int tt = 0; tt < 2; ++tt) {
      const unsigned short* kr = krow + (size_t)(kb + tt * 16) * HDIM;
      #pragma unroll
      for (int ks = 0; ks < 4; ++ks) {
        BfOp kv;
        int base = 32 * ks + (hi ? 8 : 0);
        kv.u[0] = *(const uint4*)(kr + base);
        kv.u[1] = *(const uint4*)(kr + base + 16);
        s[tt] = __builtin_amdgcn_wmma_f32_16x16x32_bf16(
            false, qa[ks].v, false, kv.v, (short)0, s[tt], false, false);
      }
    }

    // ---- online softmax (row spread across a 16-lane half) ----
    float sc[8];
    #pragma unroll
    for (int r = 0; r < 8; ++r) {
      const int qg  = q0 + r + hi * 8;
      const int kv0 = kb + l15;
      float s0 = (kv0      <= qg) ? s[0][r] : FLASH_NEG;
      float s1 = (kv0 + 16 <= qg) ? s[1][r] : FLASH_NEG;
      float mt = fmaxf(s0, s1);
      #pragma unroll
      for (int msk = 1; msk < 16; msk <<= 1) mt = fmaxf(mt, __shfl_xor(mt, msk, 32));
      float mnew  = fmaxf(mrun[r], mt);
      float scale = __expf(mrun[r] - mnew);
      float p0 = __expf(s0 - mnew);
      float p1 = __expf(s1 - mnew);
      float rs = p0 + p1;
      #pragma unroll
      for (int msk = 1; msk < 16; msk <<= 1) rs += __shfl_xor(rs, msk, 32);
      lrun[r] = lrun[r] * scale + rs;
      mrun[r] = mnew;
      sc[r]   = scale;

      int ro = r + hi * 8;
      Plds[wave][ro][l15]      = f2bf(p0);
      Plds[wave][ro][l15 + 16] = f2bf(p1);
    }

    // Rescale running output accumulators.
    #pragma unroll
    for (int nt = 0; nt < 8; ++nt) {
      #pragma unroll
      for (int r = 0; r < 8; ++r) O[nt][r] *= sc[r];
    }

    __builtin_amdgcn_wave_barrier();   // keep DS store->load in program order

    // Re-shape P (C-layout) into A-operand layout via wave-private LDS.
    BfOp pa;
    {
      const unsigned short* pp = &Plds[wave][l15][hi ? 8 : 0];
      pa.u[0] = *(const uint4*)(pp);
      pa.u[1] = *(const uint4*)(pp + 16);
    }

    // ---- O += P @ V : 8 d-tiles of 16, K = 32 keys ----
    #pragma unroll
    for (int nt = 0; nt < 8; ++nt) {
      const unsigned short* vr = vrow + (size_t)(nt * 16) * SEQ + kb + (hi ? 8 : 0);
      BfOp vb;
      vb.u[0] = *(const uint4*)(vr);
      vb.u[1] = *(const uint4*)(vr + 16);
      O[nt] = __builtin_amdgcn_wmma_f32_16x16x32_bf16(
          false, pa.v, false, vb.v, (short)0, O[nt], false, false);
    }
    __builtin_amdgcn_wave_barrier();
  }

  // ---- finalize: O / l -> bf16 [b][s][h*128 + d] ----
  #pragma unroll
  for (int nt = 0; nt < 8; ++nt) {
    #pragma unroll
    for (int r = 0; r < 8; ++r) {
      int sq = q0 + r + hi * 8;
      Out[(size_t)(b * SEQ + sq) * (NHQ * HDIM) + h * HDIM + nt * 16 + l15] =
          f2bf(O[nt][r] / lrun[r]);
    }
  }
}

// ---------------------------------------------------------------------------
// Launch
// ---------------------------------------------------------------------------
extern "C" void kernel_launch(void* const* d_in, const int* in_sizes, int n_in,
                              void* d_out, int out_size, void* d_ws, size_t ws_size,
                              hipStream_t stream)
{
  const float* hidden    = (const float*)d_in[0];
  const float* mu        = (const float*)d_in[1];
  const int*   positions = (const int*)d_in[2];
  const float* Wq        = (const float*)d_in[3];
  const float* Wk        = (const float*)d_in[4];
  const float* Wv        = (const float*)d_in[5];
  const float* Wo        = (const float*)d_in[6];
  const float* Wmu_q     = (const float*)d_in[7];
  const float* Wmu_k     = (const float*)d_in[8];
  const float* Wmu_v     = (const float*)d_in[9];
  const float* qn_w      = (const float*)d_in[10];
  const float* kn_w      = (const float*)d_in[11];
  float* out = (float*)d_out;

  char* ws = (char*)d_ws;
  size_t off = 0;
  auto alloc = [&](size_t bytes) { char* p = ws + off; off += bytes; return p; };

  float*          q_f    = (float*)         alloc(33554432);  // [4096][2048] f32
  float*          k_f    = (float*)         alloc(8388608);   // [4096][512]  f32
  float*          v_f    = (float*)         alloc(8388608);   // [4096][512]  f32
  unsigned short* q_bf   = (unsigned short*)alloc(16777216);  // [b][16][S][128]
  unsigned short* k_bf   = (unsigned short*)alloc(4194304);   // [b][4][S][128]
  unsigned short* v_bf   = (unsigned short*)alloc(4194304);   // [b][4][128][S]
  unsigned short* at_bf  = (unsigned short*)alloc(16777216);  // [4096][2048] bf16
  unsigned short* hid_bf = (unsigned short*)alloc(16777216);
  unsigned short* mu_bf  = (unsigned short*)alloc(16777216);
  unsigned short* wq_bf  = (unsigned short*)alloc(8388608);
  unsigned short* wk_bf  = (unsigned short*)alloc(2097152);
  unsigned short* wv_bf  = (unsigned short*)alloc(2097152);
  unsigned short* wo_bf  = (unsigned short*)alloc(8388608);
  unsigned short* wmq_bf = (unsigned short*)alloc(8388608);
  unsigned short* wmk_bf = (unsigned short*)alloc(2097152);
  unsigned short* wmv_bf = (unsigned short*)alloc(2097152);

  const dim3 blk256(256);

  // One-time bf16 conversions (activations + weights).
  to_bf16_kernel<<<dim3(8192), blk256, 0, stream>>>(hidden, hid_bf, 8388608);
  to_bf16_kernel<<<dim3(8192), blk256, 0, stream>>>(mu,     mu_bf,  8388608);
  to_bf16_kernel<<<dim3(4096), blk256, 0, stream>>>(Wq,     wq_bf,  4194304);
  to_bf16_kernel<<<dim3(1024), blk256, 0, stream>>>(Wk,     wk_bf,  1048576);
  to_bf16_kernel<<<dim3(1024), blk256, 0, stream>>>(Wv,     wv_bf,  1048576);
  to_bf16_kernel<<<dim3(4096), blk256, 0, stream>>>(Wo,     wo_bf,  4194304);
  to_bf16_kernel<<<dim3(4096), blk256, 0, stream>>>(Wmu_q,  wmq_bf, 4194304);
  to_bf16_kernel<<<dim3(1024), blk256, 0, stream>>>(Wmu_k,  wmk_bf, 1048576);
  to_bf16_kernel<<<dim3(1024), blk256, 0, stream>>>(Wmu_v,  wmv_bf, 1048576);

  // Fused dual-K QKV projections (async-LDS double-buffered WMMA GEMM).
  gemm_bf16_async<<<dim3(8, 64), blk256, 0, stream>>>(hid_bf, wq_bf, mu_bf, wmq_bf, q_f, 4096, 2048, 2048);
  gemm_bf16_async<<<dim3(2, 64), blk256, 0, stream>>>(hid_bf, wk_bf, mu_bf, wmk_bf, k_f, 4096, 512, 2048);
  gemm_bf16_async<<<dim3(2, 64), blk256, 0, stream>>>(hid_bf, wv_bf, mu_bf, wmv_bf, v_f, 4096, 512, 2048);

  // RMSNorm + RoPE (+ fold 1/sqrt(128) into Q); transpose to [b][h][s][d].
  norm_rope_kernel<<<dim3(65536 / 8), blk256, 0, stream>>>(q_f, qn_w, positions, q_bf, NHQ, 0.08838834764831845f);
  norm_rope_kernel<<<dim3(16384 / 8), blk256, 0, stream>>>(k_f, kn_w, positions, k_bf, NKVH, 1.0f);

  // V -> bf16 [b][hkv][d][s].
  v_pack_kernel<<<dim3(2097152 / 256), blk256, 0, stream>>>(v_f, v_bf);

  // Causal flash attention -> bf16 [b][s][h*d].
  flash_kernel<<<dim3(SEQ / 64, NHQ, 2), dim3(128), 0, stream>>>(q_bf, k_bf, v_bf, at_bf);

  // Output projection.
  gemm_bf16_async<<<dim3(8, 64), blk256, 0, stream>>>(at_bf, wo_bf, nullptr, nullptr, out, 4096, 2048, 2048);
}
